// MultiHeadAttention_56324201120277
// MI455X (gfx1250) — matrix operand order repaired and checked by reading, not compiled, not run on hardware
//
#include <hip/hip_runtime.h>
#include <hip/hip_bf16.h>
#include <stdint.h>

#define B_    2
#define N_    2048
#define DIN   1024
#define DOUT  1024
#define H_    16
#define HD_   64
#define MTOK  (B_ * N_)   // 4096

typedef __attribute__((ext_vector_type(16))) __bf16 v16bf;
typedef __attribute__((ext_vector_type(8)))  float  v8f;

union Frag {
    v16bf    v;
    uint32_t u[8];
};

__device__ __forceinline__ uint16_t f2bf(float f) {
    union { float f; uint32_t u; } a; a.f = f;
    uint32_t u = a.u;
    return (uint16_t)((u + 0x7FFFu + ((u >> 16) & 1u)) >> 16);
}
__device__ __forceinline__ uint32_t pack2(float lo, float hi) {
    return (uint32_t)f2bf(lo) | ((uint32_t)f2bf(hi) << 16);
}

__device__ __forceinline__ v8f wmma_bf16(const Frag& a, const Frag& b, v8f c) {
    return __builtin_amdgcn_wmma_f32_16x16x32_bf16(
        false, a.v, false, b.v, (short)0, c, false, false);
}

// XOR-reduction within each 16-lane half via ds_swizzle (group-of-32 mode:
// offset = xor<<10 | or<<5 | and ; and=31 keeps lane, xor in {1,2,4,8}).
__device__ __forceinline__ float row_max16(float x) {
    x = fmaxf(x, __int_as_float(__builtin_amdgcn_ds_swizzle(__float_as_int(x), 0x041F)));
    x = fmaxf(x, __int_as_float(__builtin_amdgcn_ds_swizzle(__float_as_int(x), 0x081F)));
    x = fmaxf(x, __int_as_float(__builtin_amdgcn_ds_swizzle(__float_as_int(x), 0x101F)));
    x = fmaxf(x, __int_as_float(__builtin_amdgcn_ds_swizzle(__float_as_int(x), 0x201F)));
    return x;
}
__device__ __forceinline__ float row_sum16(float x) {
    x += __int_as_float(__builtin_amdgcn_ds_swizzle(__float_as_int(x), 0x041F));
    x += __int_as_float(__builtin_amdgcn_ds_swizzle(__float_as_int(x), 0x081F));
    x += __int_as_float(__builtin_amdgcn_ds_swizzle(__float_as_int(x), 0x101F));
    x += __int_as_float(__builtin_amdgcn_ds_swizzle(__float_as_int(x), 0x201F));
    return x;
}

// ---- fragment loaders (gfx1250 wave32 WMMA, contiguous k = 16*(lane/16) + s rather than the ISA table order) ----

// A 16x32 bf16 from row-major bf16 (row stride ld elems), tile at (row0,k0).
// lane<16: row=lane, K=0..15 ; lane>=16: row=lane-16, K=16..31
__device__ __forceinline__ Frag load_a_rm(const uint16_t* src, int row0, int k0,
                                          int ld, int lane) {
    Frag f;
    int m = lane & 15, kh = lane >> 4;
    const uint32_t* p = (const uint32_t*)src;
    int base = ((row0 + m) * ld + k0 + 16 * kh) >> 1;   // even by construction
#pragma unroll
    for (int v = 0; v < 8; ++v) {
        int ksub = 2 * v;
        f.u[v] = p[base + (ksub >> 1)];
    }
    return f;
}

// B 32x16 bf16 from pre-swizzled weights: dword[(k>>1)*Nn + n] = {W[k][n],W[k+1][n]}
// B layout: VGPR v holds K=2v,2v+1 (+16 for upper lane half), lanes = N.
__device__ __forceinline__ Frag load_b_sw(const uint32_t* wsw, int kk, int col0,
                                          int Nn, int lane) {
    Frag f;
    int n = lane & 15, kh = lane >> 4;
    int base = (kk / 2 + kh * 8) * Nn + col0 + n;
#pragma unroll
    for (int v = 0; v < 8; ++v) f.u[v] = wsw[base + v * Nn];
    return f;
}

// B operand for S = Q*K^T from LDS K-tile (row-major [key][d], stride ST):
// B[k_d][n_key] = Ktile[keyoff+n][d0+k_d]; K-pairs contiguous along d.
__device__ __forceinline__ Frag load_b_kt(const uint16_t* sK, int ST, int keyoff,
                                          int d0, int lane) {
    Frag f;
    int n = lane & 15, kh = lane >> 4;
    const uint32_t* p = (const uint32_t*)sK;
    int base = ((keyoff + n) * ST + d0 + 16 * kh) >> 1;
#pragma unroll
    for (int v = 0; v < 8; ++v) f.u[v] = p[base + v];
    return f;
}

// B operand for P*V from key-pair-packed LDS V-tile:
// sVp[kp*64 + col] = {V[2kp][col], V[2kp+1][col]}  (kp = k/2, k = key in tile)
__device__ __forceinline__ Frag load_b_vp(const uint32_t* sVp, int d0, int lane) {
    Frag f;
    int n = lane & 15, kh = lane >> 4;
    int base = (8 * kh) * 64 + d0 + n;   // kp = 8*kh + v
#pragma unroll
    for (int v = 0; v < 8; ++v) f.u[v] = sVp[base + v * 64];
    return f;
}

// A operand for P (16x32) from per-wave LDS staging buffer, row stride 34.
__device__ __forceinline__ Frag load_a_p(const uint16_t* sp, int lane) {
    Frag f;
    int m = lane & 15, kh = lane >> 4;
    const uint32_t* p = (const uint32_t*)sp;
    int base = (m * 34 + 16 * kh) >> 1;
#pragma unroll
    for (int v = 0; v < 8; ++v) {
        int ksub = 2 * v;
        f.u[v] = p[base + (ksub >> 1)];
    }
    return f;
}

// ------------------------------- conversion ---------------------------------

__global__ void k_f32_to_bf16(const float* __restrict__ in,
                              uint32_t* __restrict__ out, int npair) {
    int i = blockIdx.x * blockDim.x + threadIdx.x;
    if (i < npair) out[i] = pack2(in[2 * i], in[2 * i + 1]);
}

// W[K][Nn] f32 -> pair-interleaved bf16 dwords: out[(k>>1)*Nn + n] = {W[k],W[k+1]}
__global__ void k_swizzle_w(const float* __restrict__ w,
                            uint32_t* __restrict__ out, int K, int Nn) {
    int i = blockIdx.x * blockDim.x + threadIdx.x;
    if (i >= (K / 2) * Nn) return;
    int n = i % Nn, kp = i / Nn;
    out[i] = pack2(w[(2 * kp) * Nn + n], w[(2 * kp + 1) * Nn + n]);
}

// ----------------------------------- GEMM -----------------------------------
// C[M,Nn] = A[M,K](bf16 row-major) @ Bsw[K,Nn](pair-swizzled bf16)
// 128 threads = 4 waves; wave computes 32 rows x 64 cols (8 accumulators).
template <bool OUTF32>
__global__ void __launch_bounds__(128, 1)
k_gemm(const uint16_t* __restrict__ A,
       const uint32_t* __restrict__ Bsw,
       void* __restrict__ C, const float* __restrict__ bias,
       int M, int Nn, int K, float alpha) {
    int lane = threadIdx.x & 31;
    int wave = threadIdx.x >> 5;
    int col0 = blockIdx.x * 64;
    int row0 = blockIdx.y * 128 + wave * 32;

    v8f acc[2][4] = {};
    for (int kk = 0; kk < K; kk += 32) {
        if (kk + 32 < K) {   // CDNA5 global_prefetch_b8 for next K tile
            __builtin_prefetch(&A[(row0 + (lane & 15)) * K + kk + 32], 0, 0);
            __builtin_prefetch(&Bsw[((kk + 32) >> 1) * Nn + col0 + (lane & 15)], 0, 0);
        }
        Frag a0 = load_a_rm(A, row0, kk, K, lane);
        Frag a1 = load_a_rm(A, row0 + 16, kk, K, lane);
#pragma unroll
        for (int t = 0; t < 4; ++t) {
            Frag b = load_b_sw(Bsw, kk, col0 + 16 * t, Nn, lane);
            acc[0][t] = wmma_bf16(a0, b, acc[0][t]);
            acc[1][t] = wmma_bf16(a1, b, acc[1][t]);
        }
    }

    int n = lane & 15, kh = lane >> 4;
#pragma unroll
    for (int half = 0; half < 2; ++half) {
        int rbase = row0 + 16 * half;
        if constexpr (OUTF32) {
            float* out = (float*)C;
#pragma unroll
            for (int t = 0; t < 4; ++t) {
                int col = col0 + 16 * t + n;
                float bv = bias ? bias[col] : 0.0f;
#pragma unroll
                for (int v = 0; v < 8; ++v)
                    out[(size_t)(rbase + v + 8 * kh) * Nn + col] = acc[half][t][v] + bv;
            }
        } else {
            uint16_t* out = (uint16_t*)C;
#pragma unroll
            for (int t = 0; t < 4; ++t) {
                int col = col0 + 16 * t + n;
#pragma unroll
                for (int v = 0; v < 8; ++v)
                    out[(size_t)(rbase + v + 8 * kh) * Nn + col] =
                        f2bf(acc[half][t][v] * alpha);
            }
        }
    }
}

// ------------------------------ flash attention ------------------------------
// grid.x = b*H + h (32), grid.y = query-tile-of-64 (32). 128 threads = 4 waves,
// wave w owns queries [qBase+16w, qBase+16w+15]. Keys in tiles of 32, causal.
// Next tile's global loads are issued before compute (LOADcnt overlap).
#define STK 66   // LDS row stride (elems) for K tile, co-prime banking

__global__ void __launch_bounds__(128, 1)
k_attn(const uint16_t* __restrict__ Qb,
       const uint16_t* __restrict__ Kb,
       const uint16_t* __restrict__ Vb,
       uint16_t* __restrict__ ctx) {
    __shared__ __align__(16) uint16_t sK[32 * STK];      // [key][d]
    __shared__ __align__(16) uint32_t sVp[16 * 64];      // [keypair][d] packed
    __shared__ __align__(16) uint16_t sP[4][16 * 34];

    int lane = threadIdx.x & 31;
    int wave = threadIdx.x >> 5;
    int b = blockIdx.x / H_, h = blockIdx.x % H_;
    int qBase = blockIdx.y * 64;
    int q0 = qBase + wave * 16;

    const uint16_t* Q  = Qb + (size_t)b * N_ * DOUT + h * HD_;  // row stride DOUT
    const uint16_t* Kp = Kb + (size_t)b * N_ * DOUT + h * HD_;
    const uint16_t* Vp = Vb + (size_t)b * N_ * DOUT + h * HD_;

    // Q fragments (16x64, already scaled by 1/sqrt(HD) in projection epilogue)
    Frag qa0 = load_a_rm(Q, q0, 0, DOUT, lane);
    Frag qa1 = load_a_rm(Q, q0, 32, DOUT, lane);

    v8f acc[4] = {};
    float mrow[8], lrow[8];
#pragma unroll
    for (int v = 0; v < 8; ++v) { mrow[v] = -INFINITY; lrow[v] = 0.0f; }

    int nJ = (qBase + 63) / 32 + 1;   // causal key-tile count for this block
    int cn = lane & 15, kh = lane >> 4;

    // staging-thread mappings
    int kr  = threadIdx.x >> 2;          // K: key row 0..31
    int kc  = (threadIdx.x & 3) * 16;    // K: col chunk of 16
    int vkp = threadIdx.x >> 3;          // V: key pair 0..15
    int vc  = (threadIdx.x & 7) * 8;     // V: col chunk of 8

    uint4 ka0, ka1, va0, va1;            // pipelined tile registers
    {   // preload tile j = 0
        const uint16_t* gk = Kp + (size_t)kr * DOUT + kc;
        ka0 = *(const uint4*)gk;
        ka1 = *(const uint4*)(gk + 8);
        const uint16_t* gv0 = Vp + (size_t)(2 * vkp) * DOUT + vc;
        const uint16_t* gv1 = Vp + (size_t)(2 * vkp + 1) * DOUT + vc;
        va0 = *(const uint4*)gv0;
        va1 = *(const uint4*)gv1;
    }

    for (int j = 0; j < nJ; ++j) {
        int key0 = 32 * j;

        __syncthreads();   // prior-iteration readers done with LDS
        {   // commit staged tile to LDS
            uint32_t* dk = (uint32_t*)(sK + kr * STK + kc);
            dk[0] = ka0.x; dk[1] = ka0.y; dk[2] = ka0.z; dk[3] = ka0.w;
            dk[4] = ka1.x; dk[5] = ka1.y; dk[6] = ka1.z; dk[7] = ka1.w;
            // interleave key pair: out[col] = {V[2kp][col], V[2kp+1][col]}
            uint4 o0, o1;
            o0.x = (va0.x & 0xFFFFu) | (va1.x << 16);
            o0.y = (va0.x >> 16) | (va1.x & 0xFFFF0000u);
            o0.z = (va0.y & 0xFFFFu) | (va1.y << 16);
            o0.w = (va0.y >> 16) | (va1.y & 0xFFFF0000u);
            o1.x = (va0.z & 0xFFFFu) | (va1.z << 16);
            o1.y = (va0.z >> 16) | (va1.z & 0xFFFF0000u);
            o1.z = (va0.w & 0xFFFFu) | (va1.w << 16);
            o1.w = (va0.w >> 16) | (va1.w & 0xFFFF0000u);
            *(uint4*)(sVp + vkp * 64 + vc)     = o0;
            *(uint4*)(sVp + vkp * 64 + vc + 4) = o1;
        }
        __syncthreads();

        if (j + 1 < nJ) {   // issue next tile's global loads (overlap compute)
            int knext = key0 + 32;
            const uint16_t* gk = Kp + (size_t)(knext + kr) * DOUT + kc;
            ka0 = *(const uint4*)gk;
            ka1 = *(const uint4*)(gk + 8);
            const uint16_t* gv0 = Vp + (size_t)(knext + 2 * vkp) * DOUT + vc;
            const uint16_t* gv1 = Vp + (size_t)(knext + 2 * vkp + 1) * DOUT + vc;
            va0 = *(const uint4*)gv0;
            va1 = *(const uint4*)gv1;
        }

        if (key0 <= q0 + 15) {
            // S = Q * K^T : two 16x16 tiles (keys 0-15, 16-31), K-dim 64
            v8f s0 = {}, s1 = {};
            {
                Frag b00 = load_b_kt(sK, STK, 0, 0, lane);
                Frag b01 = load_b_kt(sK, STK, 0, 32, lane);
                s0 = wmma_bf16(qa0, b00, s0);
                s0 = wmma_bf16(qa1, b01, s0);
                Frag b10 = load_b_kt(sK, STK, 16, 0, lane);
                Frag b11 = load_b_kt(sK, STK, 16, 32, lane);
                s1 = wmma_bf16(qa0, b10, s1);
                s1 = wmma_bf16(qa1, b11, s1);
            }

            // causal mask + online softmax (row = v + 8*kh, col = lane&15)
            float alpha_[8];
            uint16_t* pw = sP[wave];
#pragma unroll
            for (int v = 0; v < 8; ++v) {
                int qrow = q0 + v + 8 * kh;
                float x0 = (key0 + cn      <= qrow) ? s0[v] : -INFINITY;
                float x1 = (key0 + 16 + cn <= qrow) ? s1[v] : -INFINITY;
                float mx = row_max16(fmaxf(x0, x1));
                float mnew = fmaxf(mrow[v], mx);
                float al = __expf(mrow[v] - mnew);
                float p0 = __expf(x0 - mnew);
                float p1 = __expf(x1 - mnew);
                float rs = row_sum16(p0 + p1);
                lrow[v] = lrow[v] * al + rs;
                mrow[v] = mnew;
                alpha_[v] = al;
                pw[(v + 8 * kh) * 34 + cn]      = f2bf(p0);
                pw[(v + 8 * kh) * 34 + 16 + cn] = f2bf(p1);
            }
            // LDS ops are in-order within a wave: gather A-layout P fragment
            Frag pf = load_a_p(pw, lane);

#pragma unroll
            for (int t = 0; t < 4; ++t)
#pragma unroll
                for (int v = 0; v < 8; ++v) acc[t][v] *= alpha_[v];

#pragma unroll
            for (int t = 0; t < 4; ++t) {
                Frag bv = load_b_vp(sVp, 16 * t, lane);
                acc[t] = wmma_bf16(pf, bv, acc[t]);
            }
        }
    }

    // epilogue: ctx = acc / l, bf16, row-major [tok][DOUT]
#pragma unroll
    for (int t = 0; t < 4; ++t) {
#pragma unroll
        for (int v = 0; v < 8; ++v) {
            int qrow = q0 + v + 8 * kh;
            int d = 16 * t + cn;
            float val = acc[t][v] / lrow[v];
            ctx[((size_t)b * N_ + qrow) * DOUT + h * HD_ + d] = f2bf(val);
        }
    }
}

// --------------------------------- launcher ---------------------------------

extern "C" void kernel_launch(void* const* d_in, const int* in_sizes, int n_in,
                              void* d_out, int out_size, void* d_ws, size_t ws_size,
                              hipStream_t stream) {
    const float* x  = (const float*)d_in[0];
    const float* Wq = (const float*)d_in[1];
    const float* Wk = (const float*)d_in[2];
    const float* Wv = (const float*)d_in[3];
    const float* Wo = (const float*)d_in[4];
    const float* bo = (const float*)d_in[5];

    char* ws = (char*)d_ws;
    size_t off = 0;
    uint16_t* xb   = (uint16_t*)(ws + off); off += (size_t)MTOK * DIN * 2;   // 8 MB
    uint32_t* WqS  = (uint32_t*)(ws + off); off += (size_t)DIN * DOUT * 2;   // 2 MB
    uint32_t* WkS  = (uint32_t*)(ws + off); off += (size_t)DIN * DOUT * 2;
    uint32_t* WvS  = (uint32_t*)(ws + off); off += (size_t)DIN * DOUT * 2;
    uint32_t* WoS  = (uint32_t*)(ws + off); off += (size_t)DOUT * DOUT * 2;
    uint16_t* Qb   = (uint16_t*)(ws + off); off += (size_t)MTOK * DOUT * 2;  // 8 MB
    uint16_t* Kb   = (uint16_t*)(ws + off); off += (size_t)MTOK * DOUT * 2;
    uint16_t* Vb   = (uint16_t*)(ws + off); off += (size_t)MTOK * DOUT * 2;
    uint16_t* ctxb = (uint16_t*)(ws + off); off += (size_t)MTOK * DOUT * 2;

    // 1) convert x to bf16 (paired dword stores)
    {
        int npair = MTOK * DIN / 2;
        k_f32_to_bf16<<<(npair + 255) / 256, 256, 0, stream>>>(x, (uint32_t*)xb, npair);
    }
    // 2) swizzle weights (bf16, K-pair-interleaved)
    {
        int n = (DIN / 2) * DOUT;
        k_swizzle_w<<<(n + 255) / 256, 256, 0, stream>>>(Wq, WqS, DIN, DOUT);
        k_swizzle_w<<<(n + 255) / 256, 256, 0, stream>>>(Wk, WkS, DIN, DOUT);
        k_swizzle_w<<<(n + 255) / 256, 256, 0, stream>>>(Wv, WvS, DIN, DOUT);
        k_swizzle_w<<<(n + 255) / 256, 256, 0, stream>>>(Wo, WoS, DOUT, DOUT);
    }
    // 3) QKV projections (softmax scale folded into Q)
    {
        dim3 g(DOUT / 64, MTOK / 128), blk(128);
        k_gemm<false><<<g, blk, 0, stream>>>(xb, WqS, Qb, nullptr, MTOK, DOUT, DIN, 0.125f);
        k_gemm<false><<<g, blk, 0, stream>>>(xb, WkS, Kb, nullptr, MTOK, DOUT, DIN, 1.0f);
        k_gemm<false><<<g, blk, 0, stream>>>(xb, WvS, Vb, nullptr, MTOK, DOUT, DIN, 1.0f);
    }
    // 4) causal flash attention
    k_attn<<<dim3(B_ * H_, N_ / 64), 128, 0, stream>>>(Qb, Kb, Vb, ctxb);
    // 5) output projection + bias (fp32 out)
    {
        dim3 g(DOUT / 64, MTOK / 128), blk(128);
        k_gemm<true><<<g, blk, 0, stream>>>(ctxb, WoS, d_out, bo, MTOK, DOUT, DOUT, 1.0f);
    }
}
